// Attention_28183575396372
// MI455X (gfx1250) — matrix operand order repaired and checked by reading, not compiled, not run on hardware
//
#include <hip/hip_runtime.h>

#define DIMF   1024
#define HEADS  16
#define DHEAD  64
#define SEQ    2048
#define BATCH  2
#define ROWS   (BATCH*SEQ)     // 4096
#define BHEADS (BATCH*HEADS)   // 32
#define SOFTCAPV 50.0f
#define INV_SOFTCAP 0.02f
#define NEGBIG -1e30f

typedef __attribute__((ext_vector_type(16))) __bf16       v16bf;
typedef __attribute__((ext_vector_type(8)))  float        v8f;
typedef __attribute__((ext_vector_type(8)))  unsigned int v8u;
typedef __attribute__((ext_vector_type(4)))  int          v4i_t;

union FragU { v16bf f; v8u u; };

// ---- CDNA5 feature probes -------------------------------------------------
#if defined(__has_builtin)
#if __has_builtin(__builtin_amdgcn_global_load_async_to_lds_b128) && \
    __has_builtin(__builtin_amdgcn_s_wait_asynccnt)
#define HAVE_ASYNC 1
#endif
#if __has_builtin(__builtin_amdgcn_update_dpp)
#define HAVE_DPP 1
#endif
#if __has_builtin(__builtin_amdgcn_tanhf)
#define HAVE_TANH 1
#endif
#endif

__device__ inline void cp16(const void* g, void* l) {
#if defined(HAVE_ASYNC)
  __builtin_amdgcn_global_load_async_to_lds_b128(
      (__attribute__((address_space(1))) v4i_t*)(void*)g,
      (__attribute__((address_space(3))) v4i_t*)l, 0, 0);
#else
  *(uint4*)l = *(const uint4*)g;
#endif
}

__device__ inline void async_wait0() {
#if defined(HAVE_ASYNC)
  __builtin_amdgcn_s_wait_asynccnt(0);
#endif
}

__device__ inline float fast_tanh(float x) {
#if defined(HAVE_TANH)
  return __builtin_amdgcn_tanhf(x);
#else
  return tanhf(x);
#endif
}

// 16-lane group reductions (lanes 0-15 / 16-31 reduce independently).
#if defined(HAVE_DPP)
#define DPPF(x, ctrl) \
  __int_as_float(__builtin_amdgcn_update_dpp(0, __float_as_int(x), ctrl, 0xF, 0xF, true))
__device__ inline float red_max16(float x) {
  x = fmaxf(x, DPPF(x, 0xB1));   // quad_perm [1,0,3,2]  (xor 1)
  x = fmaxf(x, DPPF(x, 0x4E));   // quad_perm [2,3,0,1]  (xor 2)
  x = fmaxf(x, DPPF(x, 0x141));  // row_half_mirror
  x = fmaxf(x, DPPF(x, 0x140));  // row_mirror
  return x;
}
__device__ inline float red_sum16(float x) {
  x += DPPF(x, 0xB1);
  x += DPPF(x, 0x4E);
  x += DPPF(x, 0x141);
  x += DPPF(x, 0x140);
  return x;
}
#else
__device__ inline float red_max16(float x) {
  x = fmaxf(x, __shfl_xor(x, 1));
  x = fmaxf(x, __shfl_xor(x, 2));
  x = fmaxf(x, __shfl_xor(x, 4));
  x = fmaxf(x, __shfl_xor(x, 8));
  return x;
}
__device__ inline float red_sum16(float x) {
  x += __shfl_xor(x, 1);
  x += __shfl_xor(x, 2);
  x += __shfl_xor(x, 4);
  x += __shfl_xor(x, 8);
  return x;
}
#endif

// ---------------------------------------------------------------------------
__device__ inline unsigned short f2bf(float x) {
  unsigned u = __float_as_uint(x);
  u += 0x7FFFu + ((u >> 16) & 1u);
  return (unsigned short)(u >> 16);
}

__device__ inline v8f wmma_bf16(v16bf a, v16bf b, v8f c) {
  return __builtin_amdgcn_wmma_f32_16x16x32_bf16(false, a, false, b, (short)0, c,
                                                 false, false);
}

// A fragment (16x32 bf16, MxK) from row-major LDS tile; ld in ushorts (even).
__device__ inline v16bf frag_a_lds(const unsigned short* base, int ld, int koff) {
  int lane = threadIdx.x & 31;
  int row  = lane & 15;
  int hi   = lane >> 4;
  const unsigned int* bu = (const unsigned int*)base;
  FragU f;
#pragma unroll
  for (int v = 0; v < 8; ++v) {
    int k = koff + ((v >> 2) << 4) + (hi << 3) + ((v & 3) << 1);
    f.u[v] = bu[(row * ld + k) >> 1];
  }
  return f.f;
}

// B fragment (32x16 bf16, KxN) where B[k][n] = src[noff+n][koff+k];
// k-pairs contiguous in src -> b128-mergeable LDS reads, no packing.
__device__ inline v16bf frag_bT_lds(const unsigned short* base, int ld,
                                    int noff, int koff) {
  int lane = threadIdx.x & 31;
  int n  = noff + (lane & 15);
  int kb = koff + ((lane >> 4) << 3);
  const unsigned int* bu = (const unsigned int*)base;
  FragU f;
#pragma unroll
  for (int v = 0; v < 8; ++v)
    f.u[v] = bu[(n * ld + kb + ((v >> 2) << 4) + ((v & 3) << 1)) >> 1];
  return f.f;
}

// ---------------------------------------------------------------- conversions
__global__ void cvt_bf16(const float* __restrict__ in,
                         unsigned short* __restrict__ out, int n) {
  int i = blockIdx.x * blockDim.x + threadIdx.x;
  if (i < n) out[i] = f2bf(in[i]);
}

// in [K][N] f32 -> out [N][K] bf16 (write-coalesced)
__global__ void cvt_bf16_T(const float* __restrict__ in,
                           unsigned short* __restrict__ out, int K, int N) {
  int idx = blockIdx.x * blockDim.x + threadIdx.x;
  if (idx >= K * N) return;
  int n = idx / K, k = idx - n * K;
  out[idx] = f2bf(in[(size_t)k * N + n]);
}

// ---------------------------------------------------------------- gates
__global__ void gates_kernel(const float* __restrict__ x,
                             const float* __restrict__ wg,
                             float* __restrict__ g) {
  int idx = blockIdx.x * blockDim.x + threadIdx.x;     // ROWS*HEADS
  int h  = idx & (HEADS - 1);
  int rn = idx >> 4;                                   // b*SEQ + n
  const float* xr = x + (size_t)rn * DIMF;
  float s = 0.f;
#pragma unroll 4
  for (int k = 0; k < DIMF; ++k) s += xr[k] * wg[k * HEADS + h];
  float sig = 1.f / (1.f + __expf(-s));
  int b = rn >> 11, n = rn & (SEQ - 1);
  g[(((b << 4) + h) << 11) + n] = sig;
}

// ---------------------------------------------------------------- GEMM
// C = A(MxK) * BT(NxK)^T. Block tile 128x64, 4 waves x 32 rows, K-step 32,
// double-buffered async LDS staging.
// MODE 0: f32 -> Cf.  MODE 1: QKV scatter (q scaled 1/8; v stored transposed).
template <int MODE>
__global__ void __launch_bounds__(128)
gemm_bf16(const unsigned short* __restrict__ A,    // [M][Kd]
          const unsigned short* __restrict__ BT,   // [N][Kd]
          int N, int Kd,
          float* __restrict__ Cf,
          unsigned short* __restrict__ qb,
          unsigned short* __restrict__ kb,
          unsigned short* __restrict__ vbT) {
  __shared__ __align__(16) unsigned short As[2][128 * 32];
  __shared__ __align__(16) unsigned short Bs[2][64 * 32];
  int tid = threadIdx.x, lane = tid & 31, wv = tid >> 5;
  int m0 = blockIdx.y * 128, n0 = blockIdx.x * 64;
  v8f acc[2][4] = {};
  const unsigned short* Ag = A  + (size_t)m0 * Kd;
  const unsigned short* Bg = BT + (size_t)n0 * Kd;

  auto stage = [&](int buf, int k0) {
#pragma unroll
    for (int i = 0; i < 4; ++i) {                 // A tile: 128 rows x 4 segs
      int idx = tid + (i << 7);
      int row = idx >> 2, seg = idx & 3;
      cp16(Ag + (size_t)row * Kd + k0 + (seg << 3),
           &As[buf][row * 32 + (seg << 3)]);
    }
#pragma unroll
    for (int i = 0; i < 2; ++i) {                 // B tile: 64 rows x 4 segs
      int idx = tid + (i << 7);
      int row = idx >> 2, seg = idx & 3;
      cp16(Bg + (size_t)row * Kd + k0 + (seg << 3),
           &Bs[buf][row * 32 + (seg << 3)]);
    }
  };

  int nk = Kd >> 5;
  stage(0, 0);
  async_wait0();
  __syncthreads();
  for (int it = 0; it < nk; ++it) {
    int cur = it & 1;
    if (it + 1 < nk) stage(cur ^ 1, (it + 1) << 5);
    v16bf fa0 = frag_a_lds(&As[cur][(wv * 32) * 32], 32, 0);
    v16bf fa1 = frag_a_lds(&As[cur][(wv * 32 + 16) * 32], 32, 0);
#pragma unroll
    for (int nt = 0; nt < 4; ++nt) {
      v16bf fb = frag_bT_lds(&Bs[cur][0], 32, nt * 16, 0);
      acc[0][nt] = wmma_bf16(fa0, fb, acc[0][nt]);
      acc[1][nt] = wmma_bf16(fa1, fb, acc[1][nt]);
    }
    async_wait0();
    __syncthreads();
  }

  int hi = lane >> 4, col0 = lane & 15;
  if (MODE == 0) {
#pragma unroll
    for (int p = 0; p < 2; ++p)
#pragma unroll
      for (int nt = 0; nt < 4; ++nt)
#pragma unroll
        for (int r = 0; r < 8; ++r) {
          int grow = m0 + wv * 32 + p * 16 + r + (hi << 3);
          int gcol = n0 + nt * 16 + col0;
          Cf[(size_t)grow * N + gcol] = acc[p][nt][r];
        }
  } else {
    // Block-uniform decode: N-tile is 64-aligned (never crosses a 1024
    // boundary) and M-tile is 128-aligned (never crosses the batch boundary).
    int s  = n0 >> 10;                  // 0=q 1=k 2=v
    int h  = (n0 >> 6) & 15;
    int b  = m0 >> 11;
    int bh = (b << 4) + h;
    int npb = (m0 & (SEQ - 1)) + wv * 32;
    if (s == 2) {                       // v transposed: [bh][d][SEQ]
      unsigned short* base = vbT + (size_t)bh * DHEAD * SEQ;
#pragma unroll
      for (int p = 0; p < 2; ++p)
#pragma unroll
        for (int nt = 0; nt < 4; ++nt)
#pragma unroll
          for (int r = 0; r < 8; ++r) {
            int np = npb + p * 16 + r + (hi << 3);
            int d  = nt * 16 + col0;
            base[(size_t)d * SEQ + np] = f2bf(acc[p][nt][r]);
          }
    } else {                            // q/k: [bh][n][d]
      unsigned short* base = (s == 0 ? qb : kb) + (size_t)bh * SEQ * DHEAD;
      float fsc = (s == 0) ? 0.125f : 1.0f;
#pragma unroll
      for (int p = 0; p < 2; ++p)
#pragma unroll
        for (int nt = 0; nt < 4; ++nt)
#pragma unroll
          for (int r = 0; r < 8; ++r) {
            int np = npb + p * 16 + r + (hi << 3);
            base[((size_t)np << 6) + nt * 16 + col0] = f2bf(acc[p][nt][r] * fsc);
          }
    }
  }
}

// ---------------------------------------------------------------- attention
// 2 waves/block, each owns a 16-query tile; shared double-buffered K/V chunks.
__global__ void __launch_bounds__(64)
attn_kernel(const unsigned short* __restrict__ qb,
            const unsigned short* __restrict__ kbuf,
            const unsigned short* __restrict__ vbufT,
            const float* __restrict__ gates,
            unsigned short* __restrict__ aout) {
  __shared__ __align__(16) unsigned short sq[32 * 64];
  __shared__ __align__(16) unsigned short sk[2][32 * 64];
  __shared__ __align__(16) unsigned short svt[2][64 * 32];
  __shared__ __align__(16) unsigned short sp[2][16 * 32];

  int tid = threadIdx.x, lane = tid & 31, wv = tid >> 5;
  int qblk = blockIdx.x;                 // 0..63 (32 queries per block)
  int bh   = blockIdx.y;                 // 0..31
  int q0   = (qblk << 5) + (wv << 4);
  const unsigned short* Q  = qb    + ((size_t)bh * SEQ + (qblk << 5)) * DHEAD;
  const unsigned short* K  = kbuf  + (size_t)bh * SEQ * DHEAD;
  const unsigned short* VT = vbufT + (size_t)bh * DHEAD * SEQ;

  auto stage_kv = [&](int buf, int c) {
#pragma unroll
    for (int i = 0; i < 4; ++i) {        // K chunk: 32 rows x 8 segs
      int idx = tid + (i << 6);
      int row = idx >> 3, seg = idx & 7;
      cp16(K + ((size_t)(c << 5) + row) * DHEAD + (seg << 3),
           &sk[buf][row * 64 + (seg << 3)]);
    }
#pragma unroll
    for (int i = 0; i < 4; ++i) {        // V^T chunk: 64 rows x 4 segs
      int idx = tid + (i << 6);
      int row = idx >> 2, seg = idx & 3;
      cp16(VT + (size_t)row * SEQ + (c << 5) + (seg << 3),
           &svt[buf][row * 32 + (seg << 3)]);
    }
  };

  // stage Q tile (32x64)
#pragma unroll
  for (int i = 0; i < 4; ++i) {
    int idx = tid + (i << 6);
    int row = idx >> 3, seg = idx & 7;
    cp16(Q + (size_t)row * DHEAD + (seg << 3), sq + row * 64 + (seg << 3));
  }
  stage_kv(0, 0);
  async_wait0();
  __syncthreads();

  v16bf qa0 = frag_a_lds(sq + (wv << 4) * 64, 64, 0);
  v16bf qa1 = frag_a_lds(sq + (wv << 4) * 64, 64, 32);

  v8f acc[4] = {};
  float mrow[8], lrow[8];
#pragma unroll
  for (int r = 0; r < 8; ++r) { mrow[r] = NEGBIG; lrow[r] = 0.f; }

  int hi = lane >> 4, col0 = lane & 15;
  int nch = qblk + 1;

  for (int c = 0; c < nch; ++c) {
    int cur = c & 1;
    if (c + 1 < nch) stage_kv(cur ^ 1, c + 1);

    // S = Q K^T : two 16-key tiles, d=64 split into two k-steps
    v8f s0 = {}, s1 = {};
    {
      v16bf b0 = frag_bT_lds(&sk[cur][0], 64, 0, 0);
      v16bf b1 = frag_bT_lds(&sk[cur][0], 64, 0, 32);
      s0 = wmma_bf16(qa0, b0, s0);
      s0 = wmma_bf16(qa1, b1, s0);
      v16bf b2 = frag_bT_lds(&sk[cur][0], 64, 16, 0);
      v16bf b3 = frag_bT_lds(&sk[cur][0], 64, 16, 32);
      s1 = wmma_bf16(qa0, b2, s1);
      s1 = wmma_bf16(qa1, b3, s1);
    }

    // softcap + causal mask + online softmax.
    // Row max reduced across the 16-lane group (DPP); row-sum kept as
    // per-lane partials (scl is group-uniform) and reduced once at the end.
    int kcol0 = (c << 5) + col0;
    int kcol1 = kcol0 + 16;
#pragma unroll
    for (int r = 0; r < 8; ++r) {
      int qrow = q0 + r + (hi << 3);
      float c0 = SOFTCAPV * fast_tanh(s0[r] * INV_SOFTCAP);
      float c1 = SOFTCAPV * fast_tanh(s1[r] * INV_SOFTCAP);
      bool msk0 = kcol0 > qrow;
      bool msk1 = kcol1 > qrow;
      float mx = red_max16(fmaxf(msk0 ? NEGBIG : c0, msk1 ? NEGBIG : c1));
      float mnew = fmaxf(mrow[r], mx);
      float scl = __expf(mrow[r] - mnew);
      float p0 = msk0 ? 0.f : __expf(c0 - mnew);
      float p1 = msk1 ? 0.f : __expf(c1 - mnew);
      lrow[r] = lrow[r] * scl + p0 + p1;
      mrow[r] = mnew;
#pragma unroll
      for (int nt = 0; nt < 4; ++nt) acc[nt][r] *= scl;
      int prow = r + (hi << 3);
      sp[wv][prow * 32 + col0]      = f2bf(p0);
      sp[wv][prow * 32 + col0 + 16] = f2bf(p1);
    }

    // O += P V : P is 16x32 A fragment; V^T gives pair-contiguous B fragments
    v16bf pa = frag_a_lds(&sp[wv][0], 32, 0);
#pragma unroll
    for (int nt = 0; nt < 4; ++nt) {
      v16bf fv = frag_bT_lds(&svt[cur][0], 32, nt * 16, 0);
      acc[nt] = wmma_bf16(pa, fv, acc[nt]);
    }

    async_wait0();
    __syncthreads();
  }

  // epilogue: reduce l, normalize, gate, write bf16 [ROWS][HEADS*DHEAD]
  float gsc[8];
#pragma unroll
  for (int r = 0; r < 8; ++r) {
    int qrow = q0 + r + (hi << 3);
    float lt = red_sum16(lrow[r]);
    gsc[r] = gates[(size_t)bh * SEQ + qrow] / lt;
  }
  int b = bh >> 4, h = bh & 15;
#pragma unroll
  for (int nt = 0; nt < 4; ++nt) {
#pragma unroll
    for (int r = 0; r < 8; ++r) {
      int grow = (b << 11) + q0 + r + (hi << 3);
      int gcol = (h << 6) + nt * 16 + col0;
      aout[(size_t)grow * (HEADS * DHEAD) + gcol] = f2bf(acc[nt][r] * gsc[r]);
    }
  }
}

// ---------------------------------------------------------------- launch
extern "C" void kernel_launch(void* const* d_in, const int* in_sizes, int n_in,
                              void* d_out, int out_size, void* d_ws, size_t ws_size,
                              hipStream_t stream) {
  (void)in_sizes; (void)n_in; (void)out_size; (void)ws_size;
  const float* x   = (const float*)d_in[0];   // (2,2048,1024)
  const float* wqk = (const float*)d_in[1];   // (1024,3072)
  const float* wg  = (const float*)d_in[2];   // (1024,16)
  const float* wo  = (const float*)d_in[3];   // (1024,1024)
  float* out = (float*)d_out;                 // (2,2048,1024)

  char* ws = (char*)d_ws;
  unsigned short* xb     = (unsigned short*)ws; ws += (size_t)ROWS * DIMF * 2;
  unsigned short* wqkvT  = (unsigned short*)ws; ws += (size_t)3 * DIMF * DIMF * 2;
  unsigned short* woutT  = (unsigned short*)ws; ws += (size_t)DIMF * DIMF * 2;
  unsigned short* qb     = (unsigned short*)ws; ws += (size_t)BHEADS * SEQ * DHEAD * 2;
  unsigned short* kb     = (unsigned short*)ws; ws += (size_t)BHEADS * SEQ * DHEAD * 2;
  unsigned short* vbT    = (unsigned short*)ws; ws += (size_t)BHEADS * SEQ * DHEAD * 2;
  unsigned short* aout   = (unsigned short*)ws; ws += (size_t)ROWS * DIMF * 2;
  float* g               = (float*)ws;          ws += (size_t)BHEADS * SEQ * 4;

  // 1) conversions (x straight; weights transposed to [N][K] bf16)
  {
    int n1 = ROWS * DIMF;
    cvt_bf16<<<(n1 + 255) / 256, 256, 0, stream>>>(x, xb, n1);
    int n2 = DIMF * 3 * DIMF;
    cvt_bf16_T<<<(n2 + 255) / 256, 256, 0, stream>>>(wqk, wqkvT, DIMF, 3 * DIMF);
    int n3 = DIMF * DIMF;
    cvt_bf16_T<<<(n3 + 255) / 256, 256, 0, stream>>>(wo, woutT, DIMF, DIMF);
  }

  // 2) gates = sigmoid(x @ w_gates)
  gates_kernel<<<(ROWS * HEADS) / 256, 256, 0, stream>>>(x, wg, g);

  // 3) QKV projection -> q(scaled)/k/v^T bf16 buffers
  gemm_bf16<1><<<dim3(3 * DIMF / 64, ROWS / 128), 128, 0, stream>>>(
      xb, wqkvT, 3 * DIMF, DIMF, nullptr, qb, kb, vbT);

  // 4) attention
  attn_kernel<<<dim3(SEQ / 32, BHEADS), 64, 0, stream>>>(qb, kb, vbT, g, aout);

  // 5) output projection -> f32 d_out
  gemm_bf16<0><<<dim3(DIMF / 64, ROWS / 128), 128, 0, stream>>>(
      aout, woutT, DIMF, DIMF, out, nullptr, nullptr, nullptr);
}
